// QFE_GCN_86457691668576
// MI455X (gfx1250) — compile-verified, matrix operand order checked by reading, not run on hardware
//
#include <hip/hip_runtime.h>
#include <hip/hip_bf16.h>

// ---------------------------------------------------------------------------
// GCN forward for MI455X (gfx1250, wave32).
//  - Dense per-node transforms: v_wmma_f32_16x16x32_bf16 with bf16x3 split
//    (hi/lo decomposition of both activations and weights) for ~fp32 accuracy.
//  - Edge aggregation: coalesced gather of h[row], global_atomic_add_f32
//    scatter into agg[col] (bandwidth/atomic bound: ~1GB of traffic/layer).
//  - Self-loop (dinv^2 * h) fused into GEMM epilogue = also initializes agg.
// ---------------------------------------------------------------------------

#define DIN    128
#define DH     64
#define DOUT   10
#define NGRAPH 512

typedef __attribute__((ext_vector_type(16))) __bf16 bf16x16;
typedef __attribute__((ext_vector_type(8)))  float  f32x8;

__device__ __forceinline__ void split_bf16(float f, __bf16& hi, __bf16& lo) {
    hi = (__bf16)f;
    lo = (__bf16)(f - (float)hi);
}

// ---------------------------------------------------------------------------
// Degree / normalization precompute:  deg = in-degree(col) + 1 (self loop),
// dinv = rsqrt(deg), norm[e] = dinv[row]*dinv[col].
// ---------------------------------------------------------------------------
__global__ __launch_bounds__(256) void k_deg_init(float* __restrict__ deg, int n) {
    int i = blockIdx.x * blockDim.x + threadIdx.x;
    if (i < n) deg[i] = 1.0f;                       // self loop
}

__global__ __launch_bounds__(256) void k_deg_edges(const long long* __restrict__ col,
                                                   float* __restrict__ deg, int ne) {
    int e = blockIdx.x * blockDim.x + threadIdx.x;
    if (e < ne) atomicAdd(&deg[col[e]], 1.0f);
}

__global__ __launch_bounds__(256) void k_dinv(float* __restrict__ d, int n) {
    int i = blockIdx.x * blockDim.x + threadIdx.x;
    if (i < n) d[i] = rsqrtf(d[i]);                 // deg >= 1 always
}

__global__ __launch_bounds__(256) void k_norm(const long long* __restrict__ row,
                                              const long long* __restrict__ col,
                                              const float* __restrict__ dinv,
                                              float* __restrict__ nrm, int ne) {
    int e = blockIdx.x * blockDim.x + threadIdx.x;
    if (e < ne) nrm[e] = dinv[row[e]] * dinv[col[e]];
}

// ---------------------------------------------------------------------------
// WMMA GEMM:  H = X @ W  (f32 in, bf16x3-split WMMA, f32 accum)
// Epilogue:   H   -> message source buffer
//             AGG -> H * dinv^2   (self-loop contribution; inits the buffer)
// Block = 256 threads = 8 waves. CT col-tiles of 16, 8/CT row-tiles of 16.
// Fragment layouts per CDNA5 ISA 7.12.2 (wave32).
// ---------------------------------------------------------------------------
template<int K, int COLS, int CT>
__global__ __launch_bounds__(256)
void k_gcn_gemm(const float* __restrict__ X, const float* __restrict__ W,
                const float* __restrict__ dinv,
                float* __restrict__ H, float* __restrict__ AGG, int nrows)
{
    constexpr int RW    = 8 / CT;       // waves along M
    constexpr int BROWS = RW * 16;      // rows per block

    const int tid  = threadIdx.x;
    const int wave = tid >> 5;
    const int lane = tid & 31;
    const int wm   = wave / CT;
    const int wn   = wave % CT;
    const int half = lane >> 4;         // K-half interleave (ISA A/B layout)
    const int l16  = lane & 15;

    const int rowTile = blockIdx.x * BROWS + wm * 16;
    const int colTile = wn * 16;
    const int rowA    = min(rowTile + l16, nrows - 1);   // clamped load row
    const int ncol    = colTile + l16;

    f32x8 c = {};

    for (int kb = 0; kb < K; kb += 32) {
        bf16x16 a_hi, a_lo, b_hi, b_lo;
        #pragma unroll
        for (int r = 0; r < 8; ++r) {
            // K offset pattern: vgpr r<4 -> K base 0, r>=4 -> K base 16,
            // +8 for lane-half, two consecutive K per vgpr.
            const int kofs = ((r & 4) << 2) + half * 8 + (r & 3) * 2;

            // ---- A fragment: row rowA, K = kb+kofs .. kb+kofs+1 ----
            const float2 av = *(const float2*)(X + (size_t)rowA * K + kb + kofs);
            __bf16 h0, l0, h1, l1;
            split_bf16(av.x, h0, l0);
            split_bf16(av.y, h1, l1);
            a_hi[2 * r] = h0; a_hi[2 * r + 1] = h1;
            a_lo[2 * r] = l0; a_lo[2 * r + 1] = l1;

            // ---- B fragment: column ncol, same K pattern ----
            float w0 = 0.0f, w1 = 0.0f;
            if (COLS >= 16 || ncol < COLS) {
                w0 = W[(size_t)(kb + kofs)     * COLS + ncol];
                w1 = W[(size_t)(kb + kofs + 1) * COLS + ncol];
            }
            __bf16 bh0, bl0, bh1, bl1;
            split_bf16(w0, bh0, bl0);
            split_bf16(w1, bh1, bl1);
            b_hi[2 * r] = bh0; b_hi[2 * r + 1] = bh1;
            b_lo[2 * r] = bl0; b_lo[2 * r + 1] = bl1;
        }
        // bf16x3: hi*hi + hi*lo + lo*hi ~ full fp32 product, f32 accumulate
        c = __builtin_amdgcn_wmma_f32_16x16x32_bf16(false, a_hi, false, b_hi,
                                                    (short)0, c, false, false);
        c = __builtin_amdgcn_wmma_f32_16x16x32_bf16(false, a_hi, false, b_lo,
                                                    (short)0, c, false, false);
        c = __builtin_amdgcn_wmma_f32_16x16x32_bf16(false, a_lo, false, b_hi,
                                                    (short)0, c, false, false);
    }

    // ---- C/D layout: element r -> row = rowTile + half*8 + r, col = ncol ----
    #pragma unroll
    for (int r = 0; r < 8; ++r) {
        const int row = rowTile + half * 8 + r;
        if (row < nrows && (COLS >= 16 || ncol < COLS)) {
            const float v  = c[r];
            const float dv = dinv[row];
            H  [(size_t)row * COLS + ncol] = v;
            AGG[(size_t)row * COLS + ncol] = v * dv * dv;   // self-loop init
        }
    }
}

// ---------------------------------------------------------------------------
// Edge aggregation: AGG[col] += H[row] * norm[e]. TPE threads per edge
// (coalesced row read, one f32 atomic per feature).
// ---------------------------------------------------------------------------
template<int C, int TPE>
__global__ __launch_bounds__(256)
void k_edge_agg(const long long* __restrict__ row, const long long* __restrict__ col,
                const float* __restrict__ nrm, const float* __restrict__ H,
                float* __restrict__ AGG, int ne)
{
    const int t = blockIdx.x * blockDim.x + threadIdx.x;
    const int e = t / TPE;
    const int c = t % TPE;
    if (e >= ne) return;
    const long long rs = row[e];
    const long long cd = col[e];
    if (c < C) {
        const float v = H[(size_t)rs * C + c] * nrm[e];
        atomicAdd(&AGG[(size_t)cd * C + c], v);
    }
}

// ---------------------------------------------------------------------------
// Bias + optional ReLU:  OUT = act(AGG + b)
// ---------------------------------------------------------------------------
template<int C, bool RELU>
__global__ __launch_bounds__(256)
void k_bias_act(const float* __restrict__ AGG, const float* __restrict__ b,
                float* __restrict__ OUT, int n)
{
    const int t = blockIdx.x * blockDim.x + threadIdx.x;
    if (t >= n * C) return;
    const int c = t % C;
    const float v = AGG[t] + b[c];
    OUT[t] = RELU ? fmaxf(v, 0.0f) : v;
}

// ---------------------------------------------------------------------------
// Pooling: scatter-mean per graph (batch is int64), then log-softmax.
// ---------------------------------------------------------------------------
__global__ __launch_bounds__(256) void k_pool_zero(float* __restrict__ pooled,
                                                   float* __restrict__ cnt) {
    const int t = blockIdx.x * blockDim.x + threadIdx.x;
    if (t < NGRAPH * DOUT) pooled[t] = 0.0f;
    if (t < NGRAPH)        cnt[t]    = 0.0f;
}

__global__ __launch_bounds__(256)
void k_pool(const float* __restrict__ P, const long long* __restrict__ batch,
            float* __restrict__ pooled, float* __restrict__ cnt, int n)
{
    const int t = blockIdx.x * blockDim.x + threadIdx.x;
    const int i = t >> 4;
    const int c = t & 15;
    if (i >= n) return;
    const long long g = batch[i];
    if (c < DOUT) atomicAdd(&pooled[g * DOUT + c], P[(size_t)i * DOUT + c]);
    if (c == 0)   atomicAdd(&cnt[g], 1.0f);
}

__global__ __launch_bounds__(256)
void k_logsoftmax(const float* __restrict__ pooled, const float* __restrict__ cnt,
                  float* __restrict__ out)
{
    const int g = blockIdx.x * blockDim.x + threadIdx.x;
    if (g >= NGRAPH) return;
    const float inv = 1.0f / fmaxf(cnt[g], 1.0f);
    float v[DOUT];
    float m = -3.4e38f;
    #pragma unroll
    for (int c = 0; c < DOUT; ++c) { v[c] = pooled[g * DOUT + c] * inv; m = fmaxf(m, v[c]); }
    float s = 0.0f;
    #pragma unroll
    for (int c = 0; c < DOUT; ++c) s += __expf(v[c] - m);
    const float l = m + __logf(s);
    #pragma unroll
    for (int c = 0; c < DOUT; ++c) out[g * DOUT + c] = v[c] - l;
}

// ---------------------------------------------------------------------------
// Host launcher
// ---------------------------------------------------------------------------
extern "C" void kernel_launch(void* const* d_in, const int* in_sizes, int n_in,
                              void* d_out, int out_size, void* d_ws, size_t ws_size,
                              hipStream_t stream)
{
    (void)n_in; (void)out_size; (void)ws_size;

    const float*     x     = (const float*)d_in[0];
    const long long* eidx  = (const long long*)d_in[1];   // int64 per reference
    const long long* batch = (const long long*)d_in[2];
    const float* W1 = (const float*)d_in[3];  const float* b1 = (const float*)d_in[4];
    const float* W2 = (const float*)d_in[5];  const float* b2 = (const float*)d_in[6];
    const float* W3 = (const float*)d_in[7];  const float* b3 = (const float*)d_in[8];
    const float* W4 = (const float*)d_in[9];  const float* b4 = (const float*)d_in[10];
    float* out = (float*)d_out;

    const int n = in_sizes[0] / DIN;      // 100000 nodes
    const int e = in_sizes[1] / 2;        // 1600000 edges
    const long long* row = eidx;          // edge_index[0] = source
    const long long* col = eidx + e;      // edge_index[1] = target

    // Workspace carve-out (256B aligned)
    char* ws = (char*)d_ws;
    size_t off = 0;
    auto carve = [&](size_t bytes) -> void* {
        void* p = ws + off;
        off = (off + bytes + 255) & ~(size_t)255;
        return p;
    };
    float* dinv   = (float*)carve((size_t)n * 4);
    float* nrm    = (float*)carve((size_t)e * 4);
    float* H      = (float*)carve((size_t)n * DH * 4);
    float* AGG    = (float*)carve((size_t)n * DH * 4);
    float* ACT    = (float*)carve((size_t)n * DH * 4);
    float* pooled = (float*)carve((size_t)NGRAPH * DOUT * 4);
    float* cnt    = (float*)carve((size_t)NGRAPH * 4);

    auto cdiv = [](long long a, long long b) { return (unsigned)((a + b - 1) / b); };

    // ---- degree / norm ----
    k_deg_init <<<cdiv(n, 256), 256, 0, stream>>>(dinv, n);
    k_deg_edges<<<cdiv(e, 256), 256, 0, stream>>>(col, dinv, e);
    k_dinv     <<<cdiv(n, 256), 256, 0, stream>>>(dinv, n);
    k_norm     <<<cdiv(e, 256), 256, 0, stream>>>(row, col, dinv, nrm, e);

    // ---- layer 1: 128 -> 64, ReLU ----
    k_gcn_gemm<DIN, DH, 4><<<cdiv(n, 32), 256, 0, stream>>>(x, W1, dinv, H, AGG, n);
    k_edge_agg<DH, 64>    <<<cdiv((long long)e * 64, 256), 256, 0, stream>>>(row, col, nrm, H, AGG, e);
    k_bias_act<DH, true>  <<<cdiv((long long)n * DH, 256), 256, 0, stream>>>(AGG, b1, ACT, n);

    // ---- layer 2: 64 -> 64, ReLU ----
    k_gcn_gemm<DH, DH, 4><<<cdiv(n, 32), 256, 0, stream>>>(ACT, W2, dinv, H, AGG, n);
    k_edge_agg<DH, 64>   <<<cdiv((long long)e * 64, 256), 256, 0, stream>>>(row, col, nrm, H, AGG, e);
    k_bias_act<DH, true> <<<cdiv((long long)n * DH, 256), 256, 0, stream>>>(AGG, b2, ACT, n);

    // ---- layer 3: 64 -> 64, ReLU ----
    k_gcn_gemm<DH, DH, 4><<<cdiv(n, 32), 256, 0, stream>>>(ACT, W3, dinv, H, AGG, n);
    k_edge_agg<DH, 64>   <<<cdiv((long long)e * 64, 256), 256, 0, stream>>>(row, col, nrm, H, AGG, e);
    k_bias_act<DH, true> <<<cdiv((long long)n * DH, 256), 256, 0, stream>>>(AGG, b3, ACT, n);

    // ---- layer 4: 64 -> 10, no activation ----
    k_gcn_gemm<DH, DOUT, 1><<<cdiv(n, 128), 256, 0, stream>>>(ACT, W4, dinv, H, AGG, n);
    k_edge_agg<DOUT, 16>   <<<cdiv((long long)e * 16, 256), 256, 0, stream>>>(row, col, nrm, H, AGG, e);
    k_bias_act<DOUT, false><<<cdiv((long long)n * DOUT, 256), 256, 0, stream>>>(AGG, b4, ACT, n);

    // ---- pooling (scatter-mean) + log-softmax ----
    k_pool_zero <<<cdiv(NGRAPH * DOUT, 256), 256, 0, stream>>>(pooled, cnt);
    k_pool      <<<cdiv((long long)n * 16, 256), 256, 0, stream>>>(ACT, batch, pooled, cnt, n);
    k_logsoftmax<<<cdiv(NGRAPH, 256), 256, 0, stream>>>(pooled, cnt, out);
}
